// GnBlock_5952824672848
// MI455X (gfx1250) — matrix operand order repaired and checked by reading, not compiled, not run on hardware
//
#include <hip/hip_runtime.h>
#include <hip/hip_bf16.h>

#define NN 200000
#define EE 400000

typedef __bf16 bf16_t;
typedef bf16_t v16bf __attribute__((ext_vector_type(16)));
typedef bf16_t v8bf  __attribute__((ext_vector_type(8)));
typedef float  v8f   __attribute__((ext_vector_type(8)));
typedef unsigned int v4u __attribute__((ext_vector_type(4)));
typedef int          v8i __attribute__((ext_vector_type(8)));
typedef int          v4i __attribute__((ext_vector_type(4)));

#define WSTR 136      // weight stage stride (272B: 16B aligned, breaks 64-bank pattern -> 2-way max)
#define WROWS 96      // weight rows staged per chunk (96*136*2 = 26112 B)
#define THREADS 128   // 4 waves x 32 rows = 128 rows per block

#if defined(__has_builtin)
#  if __has_builtin(__builtin_amdgcn_tensor_load_to_lds)
#    define HAVE_TDM 1
#  endif
#endif

__device__ __forceinline__ unsigned short f2bf_bits(float f) {
    union { bf16_t b; unsigned short u; } c;
    c.b = (bf16_t)f;            // native cvt, RNE
    return c.u;
}

__global__ void cvt4_kernel(const float* __restrict__ src, unsigned short* __restrict__ dst,
                            long long n /* multiple of 4 */) {
    long long i = ((long long)blockIdx.x * 256 + threadIdx.x) * 4;
    if (i < n) {
        float4 v = *(const float4*)(src + i);
        ushort4 o;
        o.x = f2bf_bits(v.x); o.y = f2bf_bits(v.y);
        o.z = f2bf_bits(v.z); o.w = f2bf_bits(v.w);
        *(ushort4*)(dst + i) = o;
    }
}

__global__ void zero_kernel(float* __restrict__ p, long long n) {
    long long i = (long long)blockIdx.x * blockDim.x + threadIdx.x;
    if (i < n) p[i] = 0.0f;
}

// agg[r][0:64] += eattr[e][0:64]; agg[s][0:64] += eattr[e][64:128]
__global__ void scatter_kernel(const float* __restrict__ eattr, const int* __restrict__ ei,
                               float* __restrict__ agg) {
    long long tid = (long long)blockIdx.x * blockDim.x + threadIdx.x;
    long long e = tid >> 6;
    int j = (int)(tid & 63);
    if (e < EE) {
        long long s = ei[e];
        long long r = ei[(long long)EE + e];
        atomicAdd(agg + r * 64 + j, eattr[e * 128 + j]);
        atomicAdd(agg + s * 64 + j, eattr[e * 128 + 64 + j]);
    }
}

// Fused 3-layer MLP (Linear+SiLU, Linear+SiLU, Linear) + LayerNorm + residual.
// bf16 activations/weights, f32 accumulate. Each wave owns 32 rows (two 16-row WMMA tiles).
// mode 0 (cell): row input = [srcA0[row,0:128] | srcA1[row,0:64]],  K1=192
// mode 1 (edge): row input = [srcA0[e,0:128] | srcA1[s,0:128] | srcA1[r,0:128]], K1=384
__global__ __launch_bounds__(THREADS)
void mlp_kernel(const unsigned short* __restrict__ srcA0,
                const unsigned short* __restrict__ srcA1,
                const int*   __restrict__ ei,
                const unsigned short* __restrict__ w1,
                const unsigned short* __restrict__ w2,
                const unsigned short* __restrict__ w3,
                const float* __restrict__ b1,
                const float* __restrict__ b2,
                const float* __restrict__ b3,
                const float* __restrict__ gamma,
                const float* __restrict__ beta,
                const float* __restrict__ resid,
                unsigned short* __restrict__ outMain,   // bf16 (xc), may be null
                float* __restrict__ outRes,             // f32 residual output, may be null
                int K1, int mode, long long nRows)
{
    __shared__ __align__(16) unsigned short sW[WROWS * WSTR];   // 26112 B weight stage
    __shared__ __align__(16) unsigned short sA[4 * 32 * 128];   // 32768 B act stage (XOR-swizzled)
    __shared__ float lnstat[4][32];                             // [wave][row]=mu, [wave][16+row]=rstd

    const int lane  = threadIdx.x & 31;
    const int wave  = threadIdx.x >> 5;
    const int rlow  = lane & 15;
    const int kbase = (lane < 16) ? 0 : 8;       // A-fragment K sub-base per ISA layout
    const int mhi   = (lane < 16) ? 0 : 8;       // C/D-fragment M sub-base
    long long rb = (long long)blockIdx.x * 128 + wave * 32;
    const bool active = rb < nRows;              // N%32==0: waves all-or-nothing
    const long long rowBase = active ? rb : (nRows - 32);
    const long long row0 = rowBase + rlow;
    const long long row1 = row0 + 16;

    long long gs0 = 0, gr0 = 0, gs1 = 0, gr1 = 0;
    if (mode == 1) {
        gs0 = ei[row0];            gr0 = ei[(long long)EE + row0];
        gs1 = ei[row1];            gr1 = ei[(long long)EE + row1];
    }

    unsigned short* aW = sA + wave * 32 * 128;

    v8f acc[16];   // [0..7] tile0 (rows 0-15), [8..15] tile1 (rows 16-31)
    auto zeroAcc = [&]() {
#pragma unroll
        for (int t = 0; t < 16; ++t) acc[t] = (v8f){0.f,0.f,0.f,0.f,0.f,0.f,0.f,0.f};
    };

    // Stage `rows` weight rows (128 bf16 each) into sW with the padded 136-element stride.
    // TDM path: one tensor_load_to_lds issued by wave 0, LDS padding reproduces the stride
    // (pad_interval=5 -> every 64 DWORDs (one 256B row); pad_amount=3 -> 4 DWORDs = 16B).
    auto stageW = [&](const unsigned short* wg, int k0, int rows) {
#ifdef HAVE_TDM
        if (wave == 0) {
            unsigned lds_addr = (unsigned)(unsigned long long)(&sW[0]);
            unsigned long long ga = (unsigned long long)(wg + (long long)k0 * 128);
            v4u g0;
            g0[0] = 1u;                                   // count=1 (valid descriptor)
            g0[1] = lds_addr;                             // lds_addr [63:32]
            g0[2] = (unsigned)(ga & 0xFFFFFFFFu);         // global_addr low
            g0[3] = (unsigned)((ga >> 32) & 0x01FFFFFFu)  // global_addr [56:32]
                    | (2u << 30);                         // type=2 (image)
            v8i g1;
            g1[0] = (int)((1u << 16)        // data_size=1 (2 bytes)
                        | (1u << 20)        // pad_enable
                        | (5u << 22)        // pad_interval: 64 DWORDs
                        | (3u << 25));      // pad_amount: 4 DWORDs
            g1[1] = (int)(128u << 16);      // tensor_dim0 = 128 (elements per row)
            g1[2] = (int)((unsigned)rows << 16);  // tensor_dim1 = rows
            g1[3] = (int)(128u << 16);      // tile_dim0 = 128
            g1[4] = rows;                   // tile_dim1 = rows (tile_dim2 = 0)
            g1[5] = 128;                    // tensor_dim0_stride = 128 elements
            g1[6] = 0;
            g1[7] = 0;
            v4i gz = (v4i){0, 0, 0, 0};
#if __clang_major__ >= 23
            v8i gz8 = (v8i){0,0,0,0,0,0,0,0};
            __builtin_amdgcn_tensor_load_to_lds(g0, g1, gz, gz, gz8, 0);
#else
            __builtin_amdgcn_tensor_load_to_lds(g0, g1, gz, gz, 0);
#endif
            __builtin_amdgcn_s_wait_tensorcnt(0);
        }
#else
        for (int idx = threadIdx.x; idx < rows * 16; idx += THREADS) {
            int rr = idx >> 4, cc = idx & 15;
            uint4 v = *((const uint4*)(wg + (long long)(k0 + rr) * 128) + cc);
            *(uint4*)((char*)sW + rr * (WSTR * 2) + cc * 16) = v;
        }
#endif
    };

    // One B-fragment feeds two wmmas; next fragment prefetched before the wmmas issue.
    auto wmmaKK = [&](const v16bf& a0, const v16bf& a1, int kk) {
        const bf16_t* wcol = (const bf16_t*)sW + (kk + kbase) * WSTR + rlow;  // lane%16 = N col of B
#pragma unroll
        for (int t = 0; t < 8; ++t) {
            v16bf bm;
#pragma unroll
            for (int i = 0; i < 8; ++i) {
                bm[i]     = wcol[i * WSTR + t * 16];
                bm[8 + i] = wcol[(16 + i) * WSTR + t * 16];
            }
            acc[t]     = __builtin_amdgcn_wmma_f32_16x16x32_bf16(
                             false, a0, false, bm, (short)0, acc[t],     false, false);
            acc[8 + t] = __builtin_amdgcn_wmma_f32_16x16x32_bf16(
                             false, a1, false, bm, (short)0, acc[8 + t], false, false);
        }
    };

    // ---------------- Layer 1: K1 -> 128, bf16 A gathered from global ----------------
    zeroAcc();
    for (int k0 = 0; k0 < K1; k0 += WROWS) {
        int SR = K1 - k0; if (SR > WROWS) SR = WROWS;
        __syncthreads();
        stageW(w1, k0, SR);
        __syncthreads();
        for (int kk = 0; kk < SR; kk += 32) {
            v16bf a0, a1;
#pragma unroll
            for (int run = 0; run < 2; ++run) {
                int fb = k0 + kk + run * 16;   // uniform region selector (128-aligned regions)
                const unsigned short *base0, *base1;
                if (mode == 0) {
                    if (fb < 128) { base0 = srcA0 + row0 * 128 + fb;        base1 = srcA0 + row1 * 128 + fb; }
                    else          { base0 = srcA1 + row0 * 64 + (fb - 128); base1 = srcA1 + row1 * 64 + (fb - 128); }
                } else {
                    if (fb < 128)      { base0 = srcA0 + row0 * 128 + fb;        base1 = srcA0 + row1 * 128 + fb; }
                    else if (fb < 256) { base0 = srcA1 + gs0 * 128 + (fb - 128); base1 = srcA1 + gs1 * 128 + (fb - 128); }
                    else               { base0 = srcA1 + gr0 * 128 + (fb - 256); base1 = srcA1 + gr1 * 128 + (fb - 256); }
                }
                v8bf r0 = *(const v8bf*)(base0 + kbase);
                v8bf r1 = *(const v8bf*)(base1 + kbase);
#pragma unroll
                for (int i = 0; i < 8; ++i) { a0[run * 8 + i] = r0[i]; a1[run * 8 + i] = r1[i]; }
            }
            wmmaKK(a0, a1, kk);
        }
    }

    // Store bias+SiLU(acc) into swizzled act buffer: elem (M, c) at M*128 + ((c/8 ^ M%16)*8 + c%8)
    auto biasSiluToA = [&](const float* bias) {
#pragma unroll
        for (int tt = 0; tt < 2; ++tt) {
#pragma unroll
            for (int t = 0; t < 8; ++t) {
                float bv = bias[t * 16 + rlow];
                int chunk = 2 * t + (rlow >> 3);
#pragma unroll
                for (int rr = 0; rr < 8; ++rr) {
                    float xv = acc[tt * 8 + t][rr] + bv;
                    xv = xv / (1.0f + __expf(-xv));       // SiLU
                    int M = rr + mhi + tt * 16;
                    int pos = chunk ^ (rr + mhi);         // == chunk ^ (M & 15)
                    aW[M * 128 + pos * 8 + (rlow & 7)] = f2bf_bits(xv);
                }
            }
        }
    };

    auto gemmFromA = [&](const unsigned short* wg) {
        for (int k0 = 0; k0 < 128; k0 += WROWS) {
            int SR = 128 - k0; if (SR > WROWS) SR = WROWS;
            __syncthreads();
            stageW(wg, k0, SR);
            __syncthreads();
            for (int kk = 0; kk < SR; kk += 32) {
                int ka = k0 + kk;
                int c0 = ka + kbase, c1 = ka + 16 + kbase;   // multiples of 8
                int p0 = ((c0 >> 3) ^ rlow) << 3;
                int p1 = ((c1 >> 3) ^ rlow) << 3;
                v8bf a0lo = *(const v8bf*)(aW + rlow * 128 + p0);
                v8bf a0hi = *(const v8bf*)(aW + rlow * 128 + p1);
                v8bf a1lo = *(const v8bf*)(aW + (16 + rlow) * 128 + p0);
                v8bf a1hi = *(const v8bf*)(aW + (16 + rlow) * 128 + p1);
                v16bf a0, a1;
#pragma unroll
                for (int i = 0; i < 8; ++i) {
                    a0[i] = a0lo[i]; a0[8 + i] = a0hi[i];
                    a1[i] = a1lo[i]; a1[8 + i] = a1hi[i];
                }
                wmmaKK(a0, a1, kk);
            }
        }
    };

    // ---------------- Layers 2 and 3 ----------------
    biasSiluToA(b1);
    zeroAcc(); gemmFromA(w2);
    biasSiluToA(b2);
    zeroAcc(); gemmFromA(w3);

    // ---------------- bias3 + LayerNorm + residual, per 16-row tile ----------------
    __syncthreads();   // all waves done with sW/lnstat region hazards
    float* lnbuf = (float*)aW;                // per-wave 16x128 f32 (act buffer is dead)
    float4 g4  = *(const float4*)(gamma + lane * 4);
    float4 be4 = *(const float4*)(beta  + lane * 4);

#pragma unroll
    for (int tt = 0; tt < 2; ++tt) {
        float s1r[8], s2r[8];
#pragma unroll
        for (int rr = 0; rr < 8; ++rr) { s1r[rr] = 0.f; s2r[rr] = 0.f; }
#pragma unroll
        for (int t = 0; t < 8; ++t) {
            float bv = b3[t * 16 + rlow];
#pragma unroll
            for (int rr = 0; rr < 8; ++rr) {
                float v = acc[tt * 8 + t][rr] + bv;
                lnbuf[(rr + mhi) * 128 + t * 16 + rlow] = v;   // transpose staging for write-out
                s1r[rr] += v; s2r[rr] += v * v;
            }
        }
        // reduce over the 16 column-lanes (stays within each half-wave)
#pragma unroll
        for (int rr = 0; rr < 8; ++rr) {
#pragma unroll
            for (int mask = 1; mask < 16; mask <<= 1) {
                s1r[rr] += __shfl_xor(s1r[rr], mask, 32);
                s2r[rr] += __shfl_xor(s2r[rr], mask, 32);
            }
        }
        if (rlow == 0) {   // lanes 0 and 16 publish rows 0-7 / 8-15
#pragma unroll
            for (int rr = 0; rr < 8; ++rr) {
                float mu  = s1r[rr] * (1.0f / 128.0f);
                float var = s2r[rr] * (1.0f / 128.0f) - mu * mu;
                lnstat[wave][rr + mhi]      = mu;
                lnstat[wave][16 + rr + mhi] = rsqrtf(var + 1e-5f);
            }
        }
#pragma unroll
        for (int row = 0; row < 16; ++row) {
            float m  = lnstat[wave][row];
            float rs = lnstat[wave][16 + row];
            float4 h4 = *(const float4*)(lnbuf + row * 128 + lane * 4);
            float4 y;
            y.x = (h4.x - m) * rs * g4.x + be4.x;
            y.y = (h4.y - m) * rs * g4.y + be4.y;
            y.z = (h4.z - m) * rs * g4.z + be4.z;
            y.w = (h4.w - m) * rs * g4.w + be4.w;
            long long gidx = (rowBase + tt * 16 + row) * 128 + lane * 4;
            if (outMain && active) {
                ushort4 o;
                o.x = f2bf_bits(y.x); o.y = f2bf_bits(y.y);
                o.z = f2bf_bits(y.z); o.w = f2bf_bits(y.w);
                *(ushort4*)(outMain + gidx) = o;
            }
            if (resid && active) {
                float4 r4 = *(const float4*)(resid + gidx);
                float4 o; o.x = y.x + r4.x; o.y = y.y + r4.y; o.z = y.z + r4.z; o.w = y.w + r4.w;
                *(float4*)(outRes + gidx) = o;
            }
        }
    }
}

extern "C" void kernel_launch(void* const* d_in, const int* in_sizes, int n_in,
                              void* d_out, int out_size, void* d_ws, size_t ws_size,
                              hipStream_t stream) {
    (void)in_sizes; (void)n_in; (void)out_size; (void)ws_size;
    const float* x       = (const float*)d_in[0];
    const float* eattr   = (const float*)d_in[1];
    const float* cb_w1   = (const float*)d_in[2];
    const float* cb_b1   = (const float*)d_in[3];
    const float* cb_w2   = (const float*)d_in[4];
    const float* cb_b2   = (const float*)d_in[5];
    const float* cb_w3   = (const float*)d_in[6];
    const float* cb_b3   = (const float*)d_in[7];
    const float* cb_g    = (const float*)d_in[8];
    const float* cb_beta = (const float*)d_in[9];
    const float* eb_w1   = (const float*)d_in[10];
    const float* eb_b1   = (const float*)d_in[11];
    const float* eb_w2   = (const float*)d_in[12];
    const float* eb_b2   = (const float*)d_in[13];
    const float* eb_w3   = (const float*)d_in[14];
    const float* eb_b3   = (const float*)d_in[15];
    const float* eb_g    = (const float*)d_in[16];
    const float* eb_beta = (const float*)d_in[17];
    const int*   ei      = (const int*)d_in[18];

    // Workspace layout (region0 aliased: f32 agg scratch first, later bf16 edge_attr):
    char* ws = (char*)d_ws;
    float*          agg_f32  = (float*)ws;
    unsigned short* eattr_bf = (unsigned short*)ws;
    size_t region0 = (size_t)EE * 128 * 2;   // 102.4MB >= N*64*4 (51.2MB)
    unsigned short* agg_bf = (unsigned short*)(ws + region0);
    unsigned short* xc_bf  = agg_bf + (size_t)NN * 64;
    unsigned short* cw1 = xc_bf + (size_t)NN * 128;
    unsigned short* cw2 = cw1 + 192 * 128;
    unsigned short* cw3 = cw2 + 128 * 128;
    unsigned short* ew1 = cw3 + 128 * 128;
    unsigned short* ew2 = ew1 + 384 * 128;
    unsigned short* ew3 = ew2 + 128 * 128;

    auto cvt = [&](const float* s, unsigned short* d, long long n) {
        cvt4_kernel<<<(int)((n / 4 + 255) / 256), 256, 0, stream>>>(s, d, n);
    };

    cvt(cb_w1, cw1, 192 * 128);
    cvt(cb_w2, cw2, 128 * 128);
    cvt(cb_w3, cw3, 128 * 128);
    cvt(eb_w1, ew1, 384 * 128);
    cvt(eb_w2, ew2, 128 * 128);
    cvt(eb_w3, ew3, 128 * 128);

    long long aggN = (long long)NN * 64;
    zero_kernel<<<(int)((aggN + 255) / 256), 256, 0, stream>>>(agg_f32, aggN);
    long long st = (long long)EE * 64;
    scatter_kernel<<<(int)((st + 255) / 256), 256, 0, stream>>>(eattr, ei, agg_f32);
    cvt(agg_f32, agg_bf, aggN);
    cvt(x, xc_bf, (long long)NN * 128);          // xc_bf initialized with bf16(x)
    cvt(eattr, eattr_bf, (long long)EE * 128);   // overwrites dead agg_f32 region

    float* out_x = (float*)d_out;
    float* out_e = out_x + (size_t)NN * 128;

    int cellBlocks = (NN + 127) / 128;   // 1563 (last block half-active, wave-clamped)
    int edgeBlocks = EE / 128;           // 3125 exact

    // CellBlock round 1 (in place): xc = MLP([xc | agg])
    mlp_kernel<<<cellBlocks, THREADS, 0, stream>>>(xc_bf, agg_bf, nullptr, cw1, cw2, cw3,
        cb_b1, cb_b2, cb_b3, cb_g, cb_beta, nullptr, xc_bf, nullptr, 192, 0, NN);
    // CellBlock round 2 (agg identical; in place): xc = MLP([xc | agg]); out_x = x + xc
    mlp_kernel<<<cellBlocks, THREADS, 0, stream>>>(xc_bf, agg_bf, nullptr, cw1, cw2, cw3,
        cb_b1, cb_b2, cb_b3, cb_g, cb_beta, x, xc_bf, out_x, 192, 0, NN);
    // EdgeBlock: out_e = eattr + MLP([eattr | xc[s] | xc[r]])
    mlp_kernel<<<edgeBlocks, THREADS, 0, stream>>>(eattr_bf, xc_bf, ei, ew1, ew2, ew3,
        eb_b1, eb_b2, eb_b3, eb_g, eb_beta, eattr, nullptr, out_e, 384, 1, EE);
}
